// GNNModel_81406810128499
// MI455X (gfx1250) — matrix operand order repaired and checked by reading, not compiled, not run on hardware
//
#include <hip/hip_runtime.h>

typedef __attribute__((ext_vector_type(16))) _Float16 v16h;
typedef __attribute__((ext_vector_type(8)))  float    v8f;

#define DIN  128
#define DH   128
#define DOUT 64

__device__ __forceinline__ void fatomic_add(float* p, float v) {
  unsafeAtomicAdd(p, v);   // -> global_atomic_add_f32 (no CAS loop)
}

// ---------------------------------------------------------------------------
// Pack W [128 x ncols] fp32 row-major into WMMA B-fragment order (f16).
// Fragment id = nt*4 + kk (nt = 16-col tile, kk = 32-K step).
// B lane layout (16-bit B 32x16): lane L holds column N = L&15;
// lanes 0-15 cover K = kk*32 + 0..7 and 16..23, lanes 16-31 cover K = kk*32 + 8..15 and 24..31,
// 16 halves per lane (matches A fragment order).
// ---------------------------------------------------------------------------
__global__ __launch_bounds__(256) void pack_w_kernel(const float* __restrict__ W,
                                                     _Float16* __restrict__ Wp,
                                                     int ncols, int nFrags) {
  int gid  = blockIdx.x * blockDim.x + threadIdx.x;
  int frag = gid >> 5;
  int lane = gid & 31;
  if (frag >= nFrags) return;
  int kk   = frag & 3;          // K/32 == 4 always (K = 128)
  int nt   = frag >> 2;
  int col  = nt * 16 + (lane & 15);
  int krow = kk * 32 + (lane >> 4) * 8;
  _Float16* o = Wp + ((size_t)frag * 32 + lane) * 16;
#pragma unroll
  for (int j = 0; j < 16; ++j)
    o[j] = (_Float16)W[(size_t)(krow + j + (j >= 8 ? 8 : 0)) * ncols + col];
}

// ---------------------------------------------------------------------------
// Degree: deg = 1 (self loop) + indegree; then dinv = rsqrt(deg) in place.
// ---------------------------------------------------------------------------
__global__ __launch_bounds__(256) void deg_init_kernel(float* deg, int n) {
  int i = blockIdx.x * blockDim.x + threadIdx.x;
  if (i < n) deg[i] = 1.0f;
}
__global__ __launch_bounds__(256) void deg_count_kernel(const int* __restrict__ dst,
                                                        float* deg, int nE) {
  int e = blockIdx.x * blockDim.x + threadIdx.x;
  if (e < nE) fatomic_add(&deg[dst[e]], 1.0f);
}
__global__ __launch_bounds__(256) void deg_rsqrt_kernel(float* deg, int n) {
  int i = blockIdx.x * blockDim.x + threadIdx.x;
  if (i < n) deg[i] = rsqrtf(deg[i]);
}

// ---------------------------------------------------------------------------
// WMMA GEMM: H[nrows x (NT*16)] = (relu?)(X[nrows x 128]) @ Wpacked
// 256 threads = 8 waves; each wave owns a 16-row strip, all NT column tiles.
// ---------------------------------------------------------------------------
template <int NT, bool RELU_IN>
__global__ __launch_bounds__(256) void gemm_wmma_kernel(const float* __restrict__ X,
                                                        const _Float16* __restrict__ Wp,
                                                        float* __restrict__ H,
                                                        int nrows) {
  const int lane    = threadIdx.x & 31;
  const int wave    = threadIdx.x >> 5;
  const int rowBase = blockIdx.x * 128 + wave * 16;
  const int mrow    = lane & 15;
  const int hi      = lane >> 4;                 // 0 or 1 (lane half)

  int arow = rowBase + mrow;
  if (arow >= nrows) arow = nrows - 1;           // clamp: safe loads, EXEC stays full
  const float* xr = X + (size_t)arow * DIN;

  v8f acc[NT];
#pragma unroll
  for (int t = 0; t < NT; ++t)
#pragma unroll
    for (int j = 0; j < 8; ++j) acc[t][j] = 0.0f;

#pragma unroll
  for (int kk = 0; kk < 4; ++kk) {
    // A fragment (16x32 f16): lanes 0-15 -> K kb+0..7 and kb+16..23 (kb=kk*32);
    // lanes 16-31 -> K kb+8..15 and kb+24..31.
    const int kb = kk * 32 + hi * 8;
    float4 q0 = *reinterpret_cast<const float4*>(xr + kb);
    float4 q1 = *reinterpret_cast<const float4*>(xr + kb + 4);
    float4 q2 = *reinterpret_cast<const float4*>(xr + kb + 16);
    float4 q3 = *reinterpret_cast<const float4*>(xr + kb + 20);
    if (RELU_IN) {
      q0.x = fmaxf(q0.x, 0.f); q0.y = fmaxf(q0.y, 0.f); q0.z = fmaxf(q0.z, 0.f); q0.w = fmaxf(q0.w, 0.f);
      q1.x = fmaxf(q1.x, 0.f); q1.y = fmaxf(q1.y, 0.f); q1.z = fmaxf(q1.z, 0.f); q1.w = fmaxf(q1.w, 0.f);
      q2.x = fmaxf(q2.x, 0.f); q2.y = fmaxf(q2.y, 0.f); q2.z = fmaxf(q2.z, 0.f); q2.w = fmaxf(q2.w, 0.f);
      q3.x = fmaxf(q3.x, 0.f); q3.y = fmaxf(q3.y, 0.f); q3.z = fmaxf(q3.z, 0.f); q3.w = fmaxf(q3.w, 0.f);
    }
    v16h a;
    a[0]  = (_Float16)q0.x; a[1]  = (_Float16)q0.y; a[2]  = (_Float16)q0.z; a[3]  = (_Float16)q0.w;
    a[4]  = (_Float16)q1.x; a[5]  = (_Float16)q1.y; a[6]  = (_Float16)q1.z; a[7]  = (_Float16)q1.w;
    a[8]  = (_Float16)q2.x; a[9]  = (_Float16)q2.y; a[10] = (_Float16)q2.z; a[11] = (_Float16)q2.w;
    a[12] = (_Float16)q3.x; a[13] = (_Float16)q3.y; a[14] = (_Float16)q3.z; a[15] = (_Float16)q3.w;

#pragma unroll
    for (int nt = 0; nt < NT; ++nt) {
      const _Float16* bp = Wp + ((size_t)(nt * 4 + kk) * 32 + lane) * 16;
      v16h b = *reinterpret_cast<const v16h*>(bp);
      acc[nt] = __builtin_amdgcn_wmma_f32_16x16x32_f16(
          false, a, false, b, (short)0, acc[nt], false, false);
    }
  }

  // C/D layout: VGPR r -> row rowBase + r (lanes 0-15) or rowBase + 8 + r (lanes 16-31)
  const int NC = NT * 16;
#pragma unroll
  for (int nt = 0; nt < NT; ++nt) {
    const int col = nt * 16 + mrow;
#pragma unroll
    for (int r = 0; r < 8; ++r) {
      const int row = rowBase + hi * 8 + r;
      if (row < nrows) H[(size_t)row * NC + col] = acc[nt][r];
    }
  }
}

// ---------------------------------------------------------------------------
// out[i][c] = bias[c] + H[i][c] * dinv[i]^2   (self-loop term + bias)
// ---------------------------------------------------------------------------
template <int C>
__global__ __launch_bounds__(256) void init_bias_kernel(const float* __restrict__ H,
                                                        const float* __restrict__ dinv,
                                                        const float* __restrict__ bias,
                                                        float* __restrict__ out, int n) {
  int id = blockIdx.x * blockDim.x + threadIdx.x;
  if (id >= n * C) return;
  int i = id / C, c = id % C;
  float di = dinv[i];
  out[id] = bias[c] + H[id] * di * di;
}

// ---------------------------------------------------------------------------
// Edge scatter: one wave per edge; out[dst] += H[src] * dinv[src]*dinv[dst]
// 128 ch: float4 per lane.  64 ch: float2 per lane.
// ---------------------------------------------------------------------------
__global__ __launch_bounds__(256) void scatter128_kernel(const int* __restrict__ esrc,
                                                         const int* __restrict__ edst,
                                                         const float* __restrict__ dinv,
                                                         const float* __restrict__ H,
                                                         float* __restrict__ out, int nE) {
  int wid  = (blockIdx.x * blockDim.x + threadIdx.x) >> 5;
  int lane = threadIdx.x & 31;
  if (wid >= nE) return;
  int s = esrc[wid], d = edst[wid];
  float nrm = dinv[s] * dinv[d];
  const float4 v = *reinterpret_cast<const float4*>(H + (size_t)s * DH + lane * 4);
  float* o = out + (size_t)d * DH + lane * 4;
  fatomic_add(o + 0, v.x * nrm);
  fatomic_add(o + 1, v.y * nrm);
  fatomic_add(o + 2, v.z * nrm);
  fatomic_add(o + 3, v.w * nrm);
}

__global__ __launch_bounds__(256) void scatter64_kernel(const int* __restrict__ esrc,
                                                        const int* __restrict__ edst,
                                                        const float* __restrict__ dinv,
                                                        const float* __restrict__ H,
                                                        float* __restrict__ out, int nE) {
  int wid  = (blockIdx.x * blockDim.x + threadIdx.x) >> 5;
  int lane = threadIdx.x & 31;
  if (wid >= nE) return;
  int s = esrc[wid], d = edst[wid];
  float nrm = dinv[s] * dinv[d];
  const float2 v = *reinterpret_cast<const float2*>(H + (size_t)s * DOUT + lane * 2);
  float* o = out + (size_t)d * DOUT + lane * 2;
  fatomic_add(o + 0, v.x * nrm);
  fatomic_add(o + 1, v.y * nrm);
}

// ---------------------------------------------------------------------------
extern "C" void kernel_launch(void* const* d_in, const int* in_sizes, int n_in,
                              void* d_out, int out_size, void* d_ws, size_t ws_size,
                              hipStream_t stream) {
  const float* x   = (const float*)d_in[0];
  const float* W1  = (const float*)d_in[1];
  const float* b1  = (const float*)d_in[2];
  const float* W2  = (const float*)d_in[3];
  const float* b2  = (const float*)d_in[4];
  const int*  eidx = (const int*)d_in[5];

  const int N = in_sizes[0] / DIN;
  const int E = in_sizes[5] / 2;
  const int* esrc = eidx;
  const int* edst = eidx + E;

  char* ws = (char*)d_ws;
  _Float16* wp1 = (_Float16*)(ws + 0);                 // 32 KB
  _Float16* wp2 = (_Float16*)(ws + 32768);             // 16 KB
  float*    deg = (float*)(ws + 65536);                // N*4 B (deg -> dinv in place)
  float*    H1  = (float*)(ws + ((size_t)1 << 20));    // N*128*4 = 51.2 MB
  float*    agg = (float*)(ws + ((size_t)54 << 20));   // N*128*4 = 51.2 MB
  float*    H2  = H1;                                  // reuse (H1 dead after scatter1)
  float*    out = (float*)d_out;

  // weight packing into WMMA B-fragment order
  pack_w_kernel<<<4, 256, 0, stream>>>(W1, wp1, DH, 32);
  pack_w_kernel<<<2, 256, 0, stream>>>(W2, wp2, DOUT, 16);

  // symmetric-normalization coefficients
  deg_init_kernel <<<(N + 255) / 256, 256, 0, stream>>>(deg, N);
  deg_count_kernel<<<(E + 255) / 256, 256, 0, stream>>>(edst, deg, E);
  deg_rsqrt_kernel<<<(N + 255) / 256, 256, 0, stream>>>(deg, N);

  const int gblocks = (N + 127) / 128;

  // layer 1: H1 = x @ W1 ; agg = b1 + self-loop + edge scatter
  gemm_wmma_kernel<8, false><<<gblocks, 256, 0, stream>>>(x, wp1, H1, N);
  init_bias_kernel<DH><<<((N * DH) + 255) / 256, 256, 0, stream>>>(H1, deg, b1, agg, N);
  scatter128_kernel<<<(E + 7) / 8, 256, 0, stream>>>(esrc, edst, deg, H1, agg, E);

  // layer 2: H2 = relu(agg) @ W2 ; out = b2 + self-loop + edge scatter
  gemm_wmma_kernel<4, true><<<gblocks, 256, 0, stream>>>(agg, wp2, H2, N);
  init_bias_kernel<DOUT><<<((N * DOUT) + 255) / 256, 256, 0, stream>>>(H2, deg, b2, out, N);
  scatter64_kernel<<<(E + 7) / 8, 256, 0, stream>>>(esrc, edst, deg, H2, out, E);
}
